// GraphAttention_49237505081496
// MI455X (gfx1250) — matrix operand order repaired and checked by reading, not compiled, not run on hardware
//
#include <hip/hip_runtime.h>
#include <hip/hip_bf16.h>

typedef __attribute__((ext_vector_type(16))) _Float16 v16h;
typedef __attribute__((ext_vector_type(8)))  float    v8f;

#define NN_ 16384
#define BB_ 64
#define LL_ 512
#define DD_ 256
#define HH_ 4
#define HD_ 64
#define SCALE_ 0.0625f   // 1/sqrt(256)

#define WMMA_F16(A, B, C) \
  __builtin_amdgcn_wmma_f32_16x16x32_f16(false, (A), false, (B), (short)0, (C), false, false)

// ---- WMMA fragment loaders (wave32; contiguous k order k = 16*half + s in both A and B) ----

// A: 16x32 f16 tile, element (m,k) = src[m*lda + k]
static __device__ __forceinline__ v16h load_A16x32(const _Float16* src, int lda) {
  int lane = threadIdx.x & 31;
  int m = lane & 15;
  int hs = lane >> 4;
  const _Float16* p = src + (size_t)m * lda;
  v16h a;
#pragma unroll
  for (int j = 0; j < 4; ++j) {
    int k = 16 * hs + 2 * j;      // VGPR 0..3: K = 16*half + 0..7
    a[2 * j]     = p[k];
    a[2 * j + 1] = p[k + 1];
    int k2 = 8 + k;               // VGPR 4..7: K = 16*half + 8..15
    a[8 + 2 * j]     = p[k2];
    a[8 + 2 * j + 1] = p[k2 + 1];
  }
  return a;
}

// B: 32x16 f16 tile from transposed storage, element (k,n) = src[n*ld + k]
// (per-lane contiguous 32B -> pairs of b128 loads)
static __device__ __forceinline__ v16h load_B32x16_tr(const _Float16* src, int ld) {
  int lane = threadIdx.x & 31;
  int n = lane & 15;
  int hs = lane >> 4;
  const _Float16* p = src + (size_t)n * ld + 16 * hs;
  v16h b;
#pragma unroll
  for (int v = 0; v < 8; ++v) {
    b[2 * v]     = p[2 * v];
    b[2 * v + 1] = p[2 * v + 1];
  }
  return b;
}

// ---- kernel 1: searchsorted ----
__global__ void k_starts(const int* __restrict__ batch, int* __restrict__ starts) {
  int b = threadIdx.x;
  if (b > BB_) return;
  if (b == BB_) { starts[BB_] = NN_; return; }
  int lo = 0, hi = NN_;
  while (lo < hi) {
    int mid = (lo + hi) >> 1;
    if (batch[mid] < b) lo = mid + 1; else hi = mid;
  }
  starts[b] = lo;
}

// ---- kernel 2: weight fp32 -> f16 ----
__global__ void k_cvt(const float* __restrict__ s, _Float16* __restrict__ d, int n) {
  int i = blockIdx.x * blockDim.x + threadIdx.x;
  if (i < n) d[i] = (_Float16)s[i];
}

// ---- kernel 3: dense pack + LayerNorm -> f16 (rows beyond roundup32(cnt) skipped) ----
__global__ void k_pack_ln(const float* __restrict__ x, const int* __restrict__ starts,
                          const float* __restrict__ ln_w, const float* __restrict__ ln_b,
                          _Float16* __restrict__ xn) {
  int row = blockIdx.x;               // b*L + l
  int b = row >> 9, l = row & 511;
  int s0v = starts[b];
  int cnt = starts[b + 1] - s0v;
  int kEnd = (cnt + 31) & ~31;
  if (l >= kEnd) return;              // never read downstream
  int n = s0v + l;
  bool valid = (l < cnt);
  int d = threadIdx.x;                // 256 threads
  float v = valid ? x[(size_t)n * DD_ + d] : 0.f;
  float s = v, s2 = v * v;
#pragma unroll
  for (int o = 16; o > 0; o >>= 1) {
    s  += __shfl_xor(s,  o, 32);
    s2 += __shfl_xor(s2, o, 32);
  }
  __shared__ float ps[8], ps2[8];
  int w = threadIdx.x >> 5, ln = threadIdx.x & 31;
  if (ln == 0) { ps[w] = s; ps2[w] = s2; }
  __syncthreads();
  if (w == 0) {
    float a  = (ln < 8) ? ps[ln]  : 0.f;
    float a2 = (ln < 8) ? ps2[ln] : 0.f;
#pragma unroll
    for (int o = 4; o > 0; o >>= 1) {
      a  += __shfl_xor(a,  o, 32);
      a2 += __shfl_xor(a2, o, 32);
    }
    if (ln == 0) { ps[0] = a; ps2[0] = a2; }
  }
  __syncthreads();
  float mu  = ps[0] * (1.f / DD_);
  float var = ps2[0] * (1.f / DD_) - mu * mu;
  float y = (v - mu) * rsqrtf(var + 1e-5f) * ln_w[d] + ln_b[d];
  xn[(size_t)row * DD_ + d] = valid ? (_Float16)y : (_Float16)0.f;
}

// ---- kernel 4: QKV GEMM. Q,K -> [B*H, L, HD]; V -> transposed [B*H, HD, L] ----
__global__ void k_qkv(const _Float16* __restrict__ xn, const _Float16* __restrict__ wq,
                      const int* __restrict__ starts,
                      _Float16* __restrict__ Qh, _Float16* __restrict__ Kh,
                      _Float16* __restrict__ Vt) {
  int w = threadIdx.x >> 5;                         // 8 waves
  int row0 = blockIdx.x * 128 + w * 16;             // 256 blocks -> 32768 rows
  int bb = row0 >> 9;                               // 128-row span stays in one graph
  int cnt = starts[bb + 1] - starts[bb];
  int kEnd = (cnt + 31) & ~31;
  if ((row0 & 511) >= kEnd) return;                 // tile fully padded: never read
  int jbase = blockIdx.y * 64;                      // 12 chunks of 64 cols
  int sel = blockIdx.y >> 2;                        // 0:Q 1:K 2:V
  int h = blockIdx.y & 3;                           // head (64 cols == one head)

  v8f acc[4] = {};
#pragma unroll
  for (int k0 = 0; k0 < DD_; k0 += 32) {
    v16h a = load_A16x32(xn + (size_t)row0 * DD_ + k0, DD_);
#pragma unroll
    for (int t = 0; t < 4; ++t) {
      v16h bm = load_B32x16_tr(wq + (size_t)(jbase + t * 16) * DD_ + k0, DD_);
      acc[t] = WMMA_F16(a, bm, acc[t]);
    }
  }
  int lane = threadIdx.x & 31;
  int nn = lane & 15, hs = lane >> 4;
  size_t base = ((size_t)(bb * HH_ + h)) * LL_ * HD_;
#pragma unroll
  for (int t = 0; t < 4; ++t)
#pragma unroll
    for (int r = 0; r < 8; ++r) {
      int ll = (row0 & 511) + 8 * hs + r;
      int hd = t * 16 + nn;
      if (sel == 2) {
        Vt[base + (size_t)hd * LL_ + ll] = (_Float16)acc[t][r];       // transposed
      } else {
        _Float16* dst = (sel == 0) ? Qh : Kh;
        dst[base + (size_t)ll * HD_ + hd] = (_Float16)acc[t][r];
      }
    }
}

// ---- kernel 5: masked flash attention per (b,h), 16 q-rows per wave ----
__global__ void k_attn(const _Float16* __restrict__ Qh, const _Float16* __restrict__ Kh,
                       const _Float16* __restrict__ Vt, const int* __restrict__ starts,
                       _Float16* __restrict__ attn) {
  int g = blockIdx.x;                  // B*H*8 = 2048 blocks, 4 waves each
  int qc = g & 7;
  int h = (g >> 3) & 3;
  int b = g >> 5;
  int w = threadIdx.x >> 5;
  int q0 = qc * 64 + w * 16;

  int cnt = starts[b + 1] - starts[b];
  if (q0 >= cnt) return;               // entire wave is padded q-rows (never gathered)
  int kEnd = (cnt + 31) & ~31;         // masked keys beyond this underflow to exactly 0

  const _Float16* Qp = Qh + ((size_t)(b * HH_ + h)) * LL_ * HD_;
  const _Float16* Kp = Kh + ((size_t)(b * HH_ + h)) * LL_ * HD_;
  const _Float16* Vp = Vt + ((size_t)(b * HH_ + h)) * LL_ * HD_;   // [HD][L]

  __shared__ _Float16 Pt[4][16 * 32];  // 1 KB per wave

  int lane = threadIdx.x & 31;
  int nn = lane & 15, hs = lane >> 4;

  v16h aq0 = load_A16x32(Qp + (size_t)q0 * HD_, HD_);
  v16h aq1 = load_A16x32(Qp + (size_t)q0 * HD_ + 32, HD_);

  v8f o[4] = {};
  float rm[8], rsum[8];
#pragma unroll
  for (int r = 0; r < 8; ++r) { rm[r] = -1e30f; rsum[r] = 0.f; }

  for (int n0 = 0; n0 < kEnd; n0 += 32) {
    v8f s0 = {}, s1 = {};
    s0 = WMMA_F16(aq0, load_B32x16_tr(Kp + (size_t)n0 * HD_, HD_), s0);
    s0 = WMMA_F16(aq1, load_B32x16_tr(Kp + (size_t)n0 * HD_ + 32, HD_), s0);
    s1 = WMMA_F16(aq0, load_B32x16_tr(Kp + (size_t)(n0 + 16) * HD_, HD_), s1);
    s1 = WMMA_F16(aq1, load_B32x16_tr(Kp + (size_t)(n0 + 16) * HD_ + 32, HD_), s1);

    bool mk0 = (n0 + nn) < cnt;        // mask from graph length (batch sorted)
    bool mk1 = (n0 + 16 + nn) < cnt;
#pragma unroll
    for (int r = 0; r < 8; ++r) {
      float v0 = mk0 ? s0[r] * SCALE_ : -6250.f;   // (-1e5)*SCALE
      float v1 = mk1 ? s1[r] * SCALE_ : -6250.f;
      float tm = fmaxf(v0, v1);
#pragma unroll
      for (int off = 1; off < 16; off <<= 1) tm = fmaxf(tm, __shfl_xor(tm, off, 32));
      float nm = fmaxf(rm[r], tm);
      float fsc = __expf(rm[r] - nm);
      float p0 = __expf(v0 - nm);
      float p1 = __expf(v1 - nm);
      float psum = p0 + p1;
#pragma unroll
      for (int off = 1; off < 16; off <<= 1) psum += __shfl_xor(psum, off, 32);
      rsum[r] = rsum[r] * fsc + psum;
      rm[r] = nm;
#pragma unroll
      for (int t = 0; t < 4; ++t) o[t][r] *= fsc;
      int mrow = 8 * hs + r;
      Pt[w][mrow * 32 + nn]      = (_Float16)p0;
      Pt[w][mrow * 32 + 16 + nn] = (_Float16)p1;
    }
    v16h ap = load_A16x32(&Pt[w][0], 32);   // same-wave LDS: DS ops in-order
#pragma unroll
    for (int t = 0; t < 4; ++t)
      o[t] = WMMA_F16(ap, load_B32x16_tr(Vp + (size_t)(t * 16) * LL_ + n0, LL_), o[t]);
  }

#pragma unroll
  for (int r = 0; r < 8; ++r) {
    int row = q0 + 8 * hs + r;
    float inv = 1.f / rsum[r];
#pragma unroll
    for (int t = 0; t < 4; ++t) {
      attn[((size_t)(b * LL_ + row)) * DD_ + h * HD_ + t * 16 + nn] =
          (_Float16)(o[t][r] * inv);
    }
  }
}

// ---- kernel 6: output projection + bias + residual + ragged gather ----
__global__ void k_proj(const _Float16* __restrict__ attn, const _Float16* __restrict__ wh,
                       const float* __restrict__ h_b, const float* __restrict__ x,
                       const int* __restrict__ starts, float* __restrict__ out) {
  int w = threadIdx.x >> 5;                      // 8 waves: 2 m-tiles x 4 n-tiles
  int mt = w >> 2, nt = w & 3;
  int row0 = blockIdx.x * 32 + mt * 16;          // 1024 blocks -> 32768 rows
  int col0 = blockIdx.y * 64 + nt * 16;          // 4 blocks -> 256 cols

  int bb = row0 >> 9;                            // 32-row span stays in one graph
  int s0v = starts[bb];
  int cnt = starts[bb + 1] - s0v;
  if ((row0 & 511) >= cnt) return;               // tile has no valid rows

  v8f acc = {};
#pragma unroll
  for (int k0 = 0; k0 < DD_; k0 += 32) {
    v16h a  = load_A16x32(attn + (size_t)row0 * DD_ + k0, DD_);
    v16h bm = load_B32x16_tr(wh + (size_t)col0 * DD_ + k0, DD_);
    acc = WMMA_F16(a, bm, acc);
  }
  int lane = threadIdx.x & 31;
  int nn = lane & 15, hs = lane >> 4;
  int col = col0 + nn;
  float bias = h_b[col];
#pragma unroll
  for (int r = 0; r < 8; ++r) {
    int ll = (row0 & 511) + 8 * hs + r;
    if (ll < cnt) {
      int node = s0v + ll;
      out[(size_t)node * DD_ + col] = acc[r] + bias + x[(size_t)node * DD_ + col];
    }
  }
}

extern "C" void kernel_launch(void* const* d_in, const int* in_sizes, int n_in,
                              void* d_out, int out_size, void* d_ws, size_t ws_size,
                              hipStream_t stream) {
  const float* x     = (const float*)d_in[0];
  const float* qkv_w = (const float*)d_in[1];
  const float* h_w   = (const float*)d_in[2];
  const float* h_b   = (const float*)d_in[3];
  const float* ln_w  = (const float*)d_in[4];
  const float* ln_b  = (const float*)d_in[5];
  const int*   batch = (const int*)d_in[6];
  float* out = (float*)d_out;

  char* ws = (char*)d_ws;
  size_t off = 0;
  auto alloc = [&](size_t bytes) {
    void* p = (void*)(ws + off);
    off += (bytes + 255) & ~(size_t)255;
    return p;
  };
  int*      starts = (int*)alloc((BB_ + 1) * sizeof(int));
  _Float16* xn     = (_Float16*)alloc((size_t)BB_ * LL_ * DD_ * sizeof(_Float16));
  _Float16* Qh     = (_Float16*)alloc((size_t)BB_ * LL_ * DD_ * sizeof(_Float16));
  _Float16* Kh     = (_Float16*)alloc((size_t)BB_ * LL_ * DD_ * sizeof(_Float16));
  _Float16* Vt     = (_Float16*)alloc((size_t)BB_ * LL_ * DD_ * sizeof(_Float16));
  _Float16* attn   = (_Float16*)alloc((size_t)BB_ * LL_ * DD_ * sizeof(_Float16));
  _Float16* wq_h   = (_Float16*)alloc((size_t)3 * DD_ * DD_ * sizeof(_Float16));
  _Float16* wh_h   = (_Float16*)alloc((size_t)DD_ * DD_ * sizeof(_Float16));

  k_starts<<<1, 128, 0, stream>>>(batch, starts);
  k_cvt<<<(3 * DD_ * DD_ + 255) / 256, 256, 0, stream>>>(qkv_w, wq_h, 3 * DD_ * DD_);
  k_cvt<<<(DD_ * DD_ + 255) / 256, 256, 0, stream>>>(h_w, wh_h, DD_ * DD_);
  k_pack_ln<<<BB_ * LL_, DD_, 0, stream>>>(x, starts, ln_w, ln_b, xn);
  k_qkv<<<dim3(256, 12), 256, 0, stream>>>(xn, wq_h, starts, Qh, Kh, Vt);
  k_attn<<<BB_ * HH_ * (LL_ / 64), 128, 0, stream>>>(Qh, Kh, Vt, starts, attn);
  k_proj<<<dim3(BB_ * LL_ / 32, DD_ / 64), 256, 0, stream>>>(attn, wh_h, h_b, x, starts, out);
}
